// GATv2_8426725834831
// MI455X (gfx1250) — matrix operand order repaired and checked by reading, not compiled, not run on hardware
//
#include <hip/hip_runtime.h>
#include <hip/hip_bf16.h>
#include <math.h>

// ---------------- constants from the reference ----------------
#define HC   256          // H*C
#define NEG_IN  0.2f      // GATv2 leaky slope
#define NEG_OUT 0.01f     // outer LeakyReLU slope

typedef __attribute__((ext_vector_type(16))) __bf16 v16bf;
typedef __attribute__((ext_vector_type(8)))  float  v8f;

union BFrag { v16bf v; unsigned int u[8]; };

__device__ __forceinline__ unsigned short f2bf(float f) {
  unsigned int u = __float_as_uint(f);
  u += 0x7FFFu + ((u >> 16) & 1u);          // round-to-nearest-even
  return (unsigned short)(u >> 16);
}
#if __has_builtin(__builtin_amdgcn_cvt_pk_bf16_f32)
typedef __bf16 bf16x2 __attribute__((ext_vector_type(2)));
__device__ __forceinline__ unsigned int pkbf(float lo, float hi) {
  union { bf16x2 v; unsigned int u; } cv;
  cv.v = __builtin_amdgcn_cvt_pk_bf16_f32(lo, hi);   // v_cvt_pk_bf16_f32
  return cv.u;
}
#elif __has_builtin(__builtin_amdgcn_perm)
// RNE in 32-bit regs; merge high halves with one v_perm_b32 (no 16-bit subreg moves)
__device__ __forceinline__ unsigned int pkbf(float lo, float hi) {
  unsigned int a = __float_as_uint(lo); a += 0x7FFFu + ((a >> 16) & 1u);
  unsigned int b = __float_as_uint(hi); b += 0x7FFFu + ((b >> 16) & 1u);
  return __builtin_amdgcn_perm(b, a, 0x07060302u);   // {b[3],b[2],a[3],a[2]}
}
#else
__device__ __forceinline__ unsigned int pkbf(float lo, float hi) {
  return (unsigned int)f2bf(lo) | ((unsigned int)f2bf(hi) << 16);
}
#endif
// order-preserving float<->uint encoding for atomicMax-based segment max
__device__ __forceinline__ unsigned int encf(float f) {
  unsigned int b = __float_as_uint(f);
  return (b & 0x80000000u) ? ~b : (b | 0x80000000u);
}
__device__ __forceinline__ float decf(unsigned int u) {
  unsigned int b = (u & 0x80000000u) ? (u ^ 0x80000000u) : ~u;
  return __uint_as_float(b);
}

// gfx1250 async global->LDS copy (ASYNCcnt-tracked DMA, bypasses VGPRs).
// ldsoff: low 32 bits of the generic shared-pointer (LDS byte offset).
__device__ __forceinline__ void async_copy_b128(unsigned ldsoff, const void* gaddr) {
  asm volatile("global_load_async_to_lds_b128 %0, %1, off"
               :: "v"(ldsoff), "v"((unsigned long long)(size_t)gaddr)
               : "memory");
}
__device__ __forceinline__ void wait_async0() {
  asm volatile("s_wait_asynccnt 0x0" ::: "memory");
}

// ---------------- utility fills ----------------
__global__ void fill_f32(float* p, float v, long long n) {
  long long i = (long long)blockIdx.x * blockDim.x + threadIdx.x;
  if (i < n) p[i] = v;
}
__global__ void fill_u32(unsigned int* p, unsigned int v, long long n) {
  long long i = (long long)blockIdx.x * blockDim.x + threadIdx.x;
  if (i < n) p[i] = v;
}

// ---------------- self-loop attr: deg + mean of incoming edge_attr ----------------
__global__ void deg_attr_kernel(const int* __restrict__ dst, const float* __restrict__ ea,
                                float* __restrict__ deg, float* __restrict__ lattr, int E) {
  long long t = (long long)blockIdx.x * blockDim.x + threadIdx.x;
  if (t >= (long long)E * 16) return;
  int e = (int)(t >> 4), d = (int)(t & 15);
  int node = dst[e];
  atomicAdd(&lattr[(size_t)node * 16 + d], ea[t]);
  if (d == 0) atomicAdd(&deg[node], 1.0f);
}
__global__ void div_deg_kernel(float* lattr, const float* deg, int N) {
  long long t = (long long)blockIdx.x * blockDim.x + threadIdx.x;
  if (t >= (long long)N * 16) return;
  lattr[t] /= fmaxf(deg[t >> 4], 1.0f);
}

// ---------------- weight pre-transpose + bf16 convert: Wt[n*K+k] = bf(W[k*Nc+n]) ----------------
__global__ void transpose_w_bf16(const float* __restrict__ W, unsigned short* __restrict__ Wt,
                                 int K, int Nc) {
  long long t = (long long)blockIdx.x * blockDim.x + threadIdx.x;
  if (t >= (long long)K * Nc) return;
  int n = (int)(t / K), k = (int)(t % K);
  Wt[t] = f2bf(W[(size_t)k * Nc + n]);
}

// ---------------- WMMA bf16 GEMM: Y = act(X[M,K] @ W[K,Nc] + b) ----------------
// block: 256 threads (8 waves). block tile 64(M) x 128(N); wave tile 32x32 (2x2 WMMA).
// Wt is pre-transposed bf16 [Nc][K]. A staged via registers (needs f32->bf16 convert);
// B staged via gfx1250 async global->LDS DMA into a double buffer, overlapped with WMMA.
__global__ __launch_bounds__(256)
void gemm_bias_act_wmma(const float* __restrict__ X, const unsigned short* __restrict__ Wt,
                        const float* __restrict__ bias, float* __restrict__ Y,
                        int M, int K, int Nc, int ldx, int ldy, int act) {
  __shared__ unsigned short As[64][32];        // row-major [m][k] bf16
  __shared__ unsigned short Bs[2][128][32];    // [n][k] bf16 (W^T tile), double-buffered
  const int tid  = threadIdx.x;
  const int lane = tid & 31, wave = tid >> 5;
  const int wm = wave & 1, wn = wave >> 1;     // 2 x 4 wave grid
  const int bM = blockIdx.x * 64, bN = blockIdx.y * 128;

  v8f acc[2][2];
  #pragma unroll
  for (int a = 0; a < 2; ++a)
    #pragma unroll
    for (int b = 0; b < 2; ++b)
      #pragma unroll
      for (int r = 0; r < 8; ++r) acc[a][b][r] = 0.0f;

  const int khalf = lane >> 4;

  // staging assignments (branch-free: clamp row, epilogue guards correctness)
  const int arow = tid >> 2, akk = (tid & 3) * 8;     // 64 rows x 32 k, 8 floats/thread
  const int brow = tid >> 1, bkk = (tid & 1) * 16;    // 128 rows x 32 k, 16 bf16/thread
  int gmA = bM + arow; if (gmA >= M) gmA = M - 1;
  const float*          aptr = X  + (size_t)gmA * ldx + akk;
  const unsigned short* bptr = Wt + (size_t)(bN + brow) * K + bkk;

  // issue async DMA for B tile 0
  {
    unsigned l0 = (unsigned)(size_t)&Bs[0][brow][bkk];
    async_copy_b128(l0,      bptr);
    async_copy_b128(l0 + 16, bptr + 8);
  }
  float4 a0 = *(const float4*)(aptr);
  float4 a1 = *(const float4*)(aptr + 4);

  int buf = 0;
  for (int k0 = 0; k0 < K; k0 += 32) {
    // commit staged A tile to LDS (packed bf16, 16B stores)
    uint4 pa;
    pa.x = pkbf(a0.x, a0.y); pa.y = pkbf(a0.z, a0.w);
    pa.z = pkbf(a1.x, a1.y); pa.w = pkbf(a1.z, a1.w);
    *(uint4*)&As[arow][akk] = pa;

    wait_async0();          // our B DMA for this tile has landed in LDS
    __syncthreads();        // ... and everyone else's too

    // kick off next tile: async B into the other buffer + A into registers
    if (k0 + 32 < K) {
      unsigned l1 = (unsigned)(size_t)&Bs[buf ^ 1][brow][bkk];
      async_copy_b128(l1,      bptr + k0 + 32);
      async_copy_b128(l1 + 16, bptr + k0 + 40);
      a0 = *(const float4*)(aptr + k0 + 32);
      a1 = *(const float4*)(aptr + k0 + 36);
    }

    // A fragments: 16x32 bf16, lanes 0-15: M=0..15 K{0..7,16..23}; lanes 16-31: K{8..15,24..31}
    BFrag af[2];
    const int mr = lane & 15;
    #pragma unroll
    for (int mt = 0; mt < 2; ++mt) {
      int row = wm * 32 + mt * 16 + mr;
      #pragma unroll
      for (int v = 0; v < 8; ++v) {
        int kb = (v < 4) ? (8 * khalf + 2 * v) : (16 + 8 * khalf + 2 * (v - 4));
        af[mt].u[v] = *(const unsigned int*)&As[row][kb];
      }
    }
    // B fragments: 32x16 bf16, lanes 0-15: K{0..7,16..23}; lanes 16-31: K{8..15,24..31}
    BFrag bf[2];
    #pragma unroll
    for (int nt = 0; nt < 2; ++nt) {
      int col = wn * 32 + nt * 16 + (lane & 15);
      #pragma unroll
      for (int v = 0; v < 8; ++v) {
        int kb = (v < 4) ? (8 * khalf + 2 * v) : (16 + 8 * khalf + 2 * (v - 4));
        bf[nt].u[v] = *(const unsigned int*)&Bs[buf][col][kb];
      }
    }
    #pragma unroll
    for (int mt = 0; mt < 2; ++mt)
      #pragma unroll
      for (int nt = 0; nt < 2; ++nt)
        acc[mt][nt] = __builtin_amdgcn_wmma_f32_16x16x32_bf16(
            false, af[mt].v, false, bf[nt].v, (short)0, acc[mt][nt], false, false);
    __syncthreads();
    buf ^= 1;
  }

  // C/D layout: lane 0-15 -> N=lane, M=r; lane 16-31 -> N=lane-16, M=8+r
  const int ncol = lane & 15;
  #pragma unroll
  for (int mt = 0; mt < 2; ++mt)
    #pragma unroll
    for (int nt = 0; nt < 2; ++nt)
      #pragma unroll
      for (int r = 0; r < 8; ++r) {
        int gm = bM + wm * 32 + mt * 16 + khalf * 8 + r;
        int gn = bN + wn * 32 + nt * 16 + ncol;
        if (gm < M) {
          float v = acc[mt][nt][r] + bias[gn];
          if (act) v = (v > 0.0f) ? v : NEG_OUT * v;
          Y[(size_t)gm * ldy + gn] = v;
        }
      }
}

// ---------------- GAT edge pass A: logits + segment max ----------------
// one wave32 per edge; 256 channels -> 8 per lane; head = chunk/2.
__global__ __launch_bounds__(256)
void gat_edge_logits(const float* __restrict__ xl, const float* __restrict__ xr,
                     const int* __restrict__ src, const int* __restrict__ dst,
                     const float* __restrict__ eattr, const float* __restrict__ lattr,
                     const float* __restrict__ We, const float* __restrict__ att,
                     float* __restrict__ logits, unsigned int* __restrict__ mmax,
                     int E, long long Ea) {
  __shared__ float WeS[16 * 256];
  __shared__ float attS[256];
  const int tid = threadIdx.x;
  for (int i = tid; i < 16 * 256; i += 256) WeS[i] = We[i];
  if (tid < 256) attS[tid] = att[tid];
  __syncthreads();

  const int lane = tid & 31, wave = tid >> 5;
  long long e = (long long)blockIdx.x * 8 + wave;
  if (e >= Ea) return;

  int s, d; const float* ea;
  if (e < E) { s = src[e]; d = dst[e]; ea = eattr + (size_t)e * 16; }
  else       { s = d = (int)(e - E);   ea = lattr + (size_t)s * 16; }

  float eav[16];
  #pragma unroll
  for (int i = 0; i < 16; ++i) eav[i] = ea[i];

  const float* xls = xl + (size_t)s * HC;
  const float* xrd = xr + (size_t)d * HC;

  // hide the next gather (gfx1250 global_prefetch_b8)
  if (e + 8 < Ea) {
    long long e2 = e + 8;
    int s2, d2;
    if (e2 < E) { s2 = src[e2]; d2 = dst[e2]; } else { s2 = d2 = (int)(e2 - E); }
    __builtin_prefetch(xl + (size_t)s2 * HC + lane * 8, 0, 0);
    __builtin_prefetch(xr + (size_t)d2 * HC + lane * 8, 0, 0);
  }

  float hacc[4] = {0.f, 0.f, 0.f, 0.f};
  #pragma unroll
  for (int j = 0; j < 8; ++j) {
    int c = j * 32 + lane;                 // head = j>>1, uniform per chunk
    float ee = 0.0f;
    #pragma unroll
    for (int dd = 0; dd < 16; ++dd) ee = fmaf(eav[dd], WeS[dd * 256 + c], ee);
    float v = xls[c] + xrd[c] + ee;
    v = (v > 0.0f) ? v : NEG_IN * v;
    hacc[j >> 1] = fmaf(v, attS[c], hacc[j >> 1]);
  }
  #pragma unroll
  for (int h = 0; h < 4; ++h)
    for (int off = 16; off > 0; off >>= 1)
      hacc[h] += __shfl_xor(hacc[h], off, 32);
  if (lane == 0) {
    #pragma unroll
    for (int h = 0; h < 4; ++h) {
      logits[e * 4 + h] = hacc[h];
      atomicMax(&mmax[(size_t)d * 4 + h], encf(hacc[h]));
    }
  }
}

// ---------------- GAT edge pass B: exp + segment sum ----------------
__global__ void gat_edge_softmax_num(const float* __restrict__ logits,
                                     const unsigned int* __restrict__ mmax,
                                     const int* __restrict__ dst,
                                     float* __restrict__ alphan, float* __restrict__ denom,
                                     int E, long long Ea) {
  long long t = (long long)blockIdx.x * blockDim.x + threadIdx.x;
  if (t >= Ea * 4) return;
  long long e = t >> 2; int h = (int)(t & 3);
  int d = (e < E) ? dst[e] : (int)(e - E);
  float ex = __expf(logits[t] - decf(mmax[(size_t)d * 4 + h]));
  alphan[t] = ex;
  atomicAdd(&denom[(size_t)d * 4 + h], ex);
}

// ---------------- GAT edge pass C: alpha-weighted scatter add ----------------
__global__ __launch_bounds__(256)
void gat_edge_aggregate(const float* __restrict__ xl, const float* __restrict__ alphan,
                        const float* __restrict__ denom,
                        const int* __restrict__ src, const int* __restrict__ dst,
                        float* __restrict__ accv, int E, long long Ea) {
  const int tid = threadIdx.x, lane = tid & 31, wave = tid >> 5;
  long long e = (long long)blockIdx.x * 8 + wave;
  if (e >= Ea) return;
  int s, d;
  if (e < E) { s = src[e]; d = dst[e]; } else { s = d = (int)(e - E); }
  float w[4];
  #pragma unroll
  for (int h = 0; h < 4; ++h) w[h] = alphan[e * 4 + h] / denom[(size_t)d * 4 + h];
  const float* xls = xl + (size_t)s * HC;
  float* ad = accv + (size_t)d * HC;
  if (e + 8 < Ea) {
    long long e2 = e + 8;
    int s2 = (e2 < E) ? src[e2] : (int)(e2 - E);
    __builtin_prefetch(xl + (size_t)s2 * HC + lane * 8, 0, 0);
  }
  #pragma unroll
  for (int j = 0; j < 8; ++j) {
    int c = j * 32 + lane;
    atomicAdd(&ad[c], xls[c] * w[j >> 1]);
  }
}

// ---------------- init acc with output bias; finalize with outer leaky ----------------
__global__ void init_acc(float* accv, const float* __restrict__ bo, long long n) {
  long long t = (long long)blockIdx.x * blockDim.x + threadIdx.x;
  if (t < n) accv[t] = bo[t & 255];
}
__global__ void finalize_leaky(const float* __restrict__ accv, float* __restrict__ hslice,
                               long long n) {
  long long t = (long long)blockIdx.x * blockDim.x + threadIdx.x;
  if (t >= n) return;
  long long row = t >> 8; int c = (int)(t & 255);
  float v = accv[t];
  v = (v > 0.0f) ? v : NEG_OUT * v;
  hslice[row * 1024 + c] = v;
}

// ---------------- global_add_pool + re-broadcast into hcat ----------------
__global__ void pool_add(const float* __restrict__ hslice, const int* __restrict__ batch,
                         float* __restrict__ pool, long long n) {
  long long t = (long long)blockIdx.x * blockDim.x + threadIdx.x;
  if (t >= n) return;
  long long row = t >> 8; int c = (int)(t & 255);
  atomicAdd(&pool[(size_t)batch[row] * 256 + c], hslice[row * 1024 + c]);
}
__global__ void pool_gather(const float* __restrict__ pool, const int* __restrict__ batch,
                            float* __restrict__ hslice, long long n) {
  long long t = (long long)blockIdx.x * blockDim.x + threadIdx.x;
  if (t >= n) return;
  long long row = t >> 8; int c = (int)(t & 255);
  hslice[row * 1024 + c] = pool[(size_t)batch[row] * 256 + c];
}

// ---------------- final 128->1 dot + sigmoid, one wave per node ----------------
__global__ __launch_bounds__(256)
void mlp_final_sigmoid(const float* __restrict__ m2, const float* __restrict__ Wf,
                       const float* __restrict__ bf, float* __restrict__ out, int N) {
  const int tid = threadIdx.x, lane = tid & 31, wave = tid >> 5;
  long long n = (long long)blockIdx.x * 8 + wave;
  if (n >= N) return;
  const float* row = m2 + (size_t)n * 128;
  float acc = 0.0f;
  #pragma unroll
  for (int j = 0; j < 4; ++j) acc = fmaf(row[j * 32 + lane], Wf[j * 32 + lane], acc);
  for (int off = 16; off > 0; off >>= 1) acc += __shfl_xor(acc, off, 32);
  if (lane == 0) out[n] = 1.0f / (1.0f + __expf(-(acc + bf[0])));
}

// =====================================================================
extern "C" void kernel_launch(void* const* d_in, const int* in_sizes, int n_in,
                              void* d_out, int out_size, void* d_ws, size_t ws_size,
                              hipStream_t stream) {
  const float* x     = (const float*)d_in[0];
  const int*   ei    = (const int*)d_in[1];
  const float* eattr = (const float*)d_in[2];
  const int*   batch = (const int*)d_in[3];
  const int N  = in_sizes[0] / 128;
  const int E  = in_sizes[1] / 2;
  const int Bg = 500;
  const long long Ea = (long long)E + N;
  const int* src = ei;
  const int* dst = ei + E;

  const float* cp[3][7];  // Wl, bl, Wr, br, We, att, bo
  for (int l = 0; l < 3; ++l)
    for (int k = 0; k < 7; ++k)
      cp[l][k] = (const float*)d_in[4 + l * 7 + k];
  const float* mW0 = (const float*)d_in[25]; const float* mb0 = (const float*)d_in[26];
  const float* mW1 = (const float*)d_in[27]; const float* mb1 = (const float*)d_in[28];
  const float* mW2 = (const float*)d_in[29]; const float* mb2 = (const float*)d_in[30];

  // workspace carve-out
  char* wsp = (char*)d_ws;
  auto alloc = [&](size_t bytes) -> void* {
    void* p = wsp; wsp += (bytes + 255) & ~(size_t)255; return p;
  };
  float* hcat   = (float*)alloc((size_t)N * 1024 * 4);  // [h1|h2|h3|pool[batch]]
  float* xl     = (float*)alloc((size_t)N * HC * 4);
  float* xr     = (float*)alloc((size_t)N * HC * 4);
  float* accv   = (float*)alloc((size_t)N * HC * 4);
  float* logits = (float*)alloc((size_t)Ea * 4 * 4);
  float* alphan = (float*)alloc((size_t)Ea * 4 * 4);
  unsigned int* mmax = (unsigned int*)alloc((size_t)N * 4 * 4);
  float* denom  = (float*)alloc((size_t)N * 4 * 4);
  float* deg    = (float*)alloc((size_t)N * 4);
  float* lattr  = (float*)alloc((size_t)N * 16 * 4);
  float* pool   = (float*)alloc((size_t)Bg * 256 * 4);
  float* m1     = (float*)alloc((size_t)N * 256 * 4);
  float* m2     = (float*)alloc((size_t)N * 128 * 4);
  unsigned short* wt = (unsigned short*)alloc((size_t)1024 * 256 * 2);  // shared bf16 W^T buffer

  auto cdiv = [](long long a, long long b) { return (unsigned)((a + b - 1) / b); };

  // self-loop edge_attr = mean of incoming edge_attr
  fill_f32<<<cdiv(N, 256), 256, 0, stream>>>(deg, 0.f, N);
  fill_f32<<<cdiv((long long)N * 16, 256), 256, 0, stream>>>(lattr, 0.f, (long long)N * 16);
  deg_attr_kernel<<<cdiv((long long)E * 16, 256), 256, 0, stream>>>(dst, eattr, deg, lattr, E);
  div_deg_kernel<<<cdiv((long long)N * 16, 256), 256, 0, stream>>>(lattr, deg, N);

  for (int l = 0; l < 3; ++l) {
    const float* Xin = (l == 0) ? x : (hcat + (size_t)(l - 1) * 256);
    const int K   = (l == 0) ? 128 : 256;
    const int ldx = (l == 0) ? 128 : 1024;
    dim3 g(cdiv(N, 64), 2);
    transpose_w_bf16<<<cdiv((long long)K * 256, 256), 256, 0, stream>>>(cp[l][0], wt, K, 256);
    gemm_bias_act_wmma<<<g, 256, 0, stream>>>(Xin, wt, cp[l][1], xl, N, K, 256, ldx, 256, 0);
    transpose_w_bf16<<<cdiv((long long)K * 256, 256), 256, 0, stream>>>(cp[l][2], wt, K, 256);
    gemm_bias_act_wmma<<<g, 256, 0, stream>>>(Xin, wt, cp[l][3], xr, N, K, 256, ldx, 256, 0);

    fill_u32<<<cdiv((long long)N * 4, 256), 256, 0, stream>>>(mmax, 0u, (long long)N * 4);
    fill_f32<<<cdiv((long long)N * 4, 256), 256, 0, stream>>>(denom, 0.f, (long long)N * 4);
    init_acc<<<cdiv((long long)N * 256, 256), 256, 0, stream>>>(accv, cp[l][6], (long long)N * 256);

    gat_edge_logits<<<cdiv(Ea, 8), 256, 0, stream>>>(xl, xr, src, dst, eattr, lattr,
                                                     cp[l][4], cp[l][5], logits, mmax, E, Ea);
    gat_edge_softmax_num<<<cdiv(Ea * 4, 256), 256, 0, stream>>>(logits, mmax, dst, alphan, denom, E, Ea);
    gat_edge_aggregate<<<cdiv(Ea, 8), 256, 0, stream>>>(xl, alphan, denom, src, dst, accv, E, Ea);
    finalize_leaky<<<cdiv((long long)N * 256, 256), 256, 0, stream>>>(accv, hcat + (size_t)l * 256,
                                                                      (long long)N * 256);
  }

  // global_add_pool over final h, then broadcast back per node
  fill_f32<<<cdiv((long long)Bg * 256, 256), 256, 0, stream>>>(pool, 0.f, (long long)Bg * 256);
  pool_add<<<cdiv((long long)N * 256, 256), 256, 0, stream>>>(hcat + 512, batch, pool, (long long)N * 256);
  pool_gather<<<cdiv((long long)N * 256, 256), 256, 0, stream>>>(pool, batch, hcat + 768, (long long)N * 256);

  // classifier MLP
  transpose_w_bf16<<<cdiv((long long)1024 * 256, 256), 256, 0, stream>>>(mW0, wt, 1024, 256);
  gemm_bias_act_wmma<<<dim3(cdiv(N, 64), 2), 256, 0, stream>>>(hcat, wt, mb0, m1, N, 1024, 256, 1024, 256, 1);
  transpose_w_bf16<<<cdiv((long long)256 * 128, 256), 256, 0, stream>>>(mW1, wt, 256, 128);
  gemm_bias_act_wmma<<<dim3(cdiv(N, 64), 1), 256, 0, stream>>>(m1, wt, mb1, m2, N, 256, 128, 256, 128, 1);
  mlp_final_sigmoid<<<cdiv(N, 8), 256, 0, stream>>>(m2, mW2, mb2, (float*)d_out, N);
}
